// ReaxFF_nn_force_70420283785965
// MI455X (gfx1250) — compile-verified, matrix operand order checked by reading, not run on hardware
//
#include <hip/hip_runtime.h>

// CDNA5 / gfx1250 ReaxFF-NN energy kernel.
// One block per batch; bond terms + per-bond MLP via v_wmma_f32_16x16x32_f16
// (weights in A-fragment, activations transposed in B-fragment so the
// inter-layer repack is lane-local; hidden-layer biases folded into the WMMA
// C operand), atom accumulators via LDS ds_add_f32. Divisions lowered to
// v_rcp_f32 via __builtin_amdgcn_rcpf. Atom-pass uniforms are loaded after
// the bond loop to reduce SGPR live range (avoids readlane spill traffic).

typedef __attribute__((ext_vector_type(16))) _Float16 v16h;
typedef __attribute__((ext_vector_type(8)))  float    v8f;

#define NATOMS_  512
#define NBONDS_  4096
#define THREADS_ 512
#define NWAVES_  (THREADS_ / 32)

__device__ __forceinline__ float fast_rcp_(float v) {
    return __builtin_amdgcn_rcpf(v);          // v_rcp_f32
}

__device__ __forceinline__ float sigmoidf_(float v) {
    return fast_rcp_(1.0f + __expf(-v));       // v_exp_f32 + v_rcp_f32
}

__device__ __forceinline__ float taperf_(float r, float rmin, float rmax) {
    float r3   = (r > rmax) ? 1.0f : 0.0f;
    bool  ok   = (r <= rmax) && (r > rmin);
    float r2   = ok ? r : 0.0f;
    float r20  = ok ? 1.0f : 0.0f;
    float dmm  = rmin - rmax;
    float rterm = fast_rcp_(dmm * dmm * dmm);
    float rm   = rmin * r20;
    float rd   = rm - r2;
    float trm1 = rm + 2.0f * r2 - 3.0f * rmax * r20;
    return rterm * rd * rd * trm1 + r3;
}

__global__ __launch_bounds__(THREADS_, 1)
void reaxff_nn_force_kernel(const float* __restrict__ x,
                            const float* __restrict__ cell,
                            const float* __restrict__ rcell,
                            const float* __restrict__ sp_p,
                            const float* __restrict__ gp,
                            const float* __restrict__ bp,
                            const float* __restrict__ fe_wi,
                            const float* __restrict__ fe_w,
                            const float* __restrict__ fe_b,
                            const float* __restrict__ fe_wo,
                            const float* __restrict__ fe_bo,
                            const int*   __restrict__ bdid,
                            const int*   __restrict__ spec,
                            float*       __restrict__ out)
{
    __shared__ float sDelta[NATOMS_];
    __shared__ float sDpi[NATOMS_];
    __shared__ float sSO[NATOMS_];
    __shared__ float sEnergy;

    const int b    = blockIdx.x;
    const int tid  = threadIdx.x;
    const int lane = tid & 31;
    const int wid  = tid >> 5;

    for (int a = tid; a < NATOMS_; a += THREADS_) {
        sDelta[a] = 0.0f; sDpi[a] = 0.0f; sSO[a] = 0.0f;
    }
    if (tid == 0) sEnergy = 0.0f;
    __syncthreads();

    // ---- uniforms needed inside the bond loop only ----
    const float botol = gp[6];
    const float rosi = bp[0], ropi = bp[1], ropp = bp[2];
    const float bo1 = bp[3], bo2 = bp[4], bo3 = bp[5], bo4 = bp[6];
    const float bo5 = bp[7], bo6 = bp[8], Desi = bp[9];
    const float bo_out = fe_bo[0];
    const float inv_rosi = fast_rcp_(rosi);
    const float inv_ropi = fast_rcp_(ropi);
    const float inv_ropp = fast_rcp_(ropp);

    float cl[9], rc[9];
#pragma unroll
    for (int k = 0; k < 9; ++k) { cl[k] = cell[b * 9 + k]; rc[k] = rcell[b * 9 + k]; }

    // ---- constant weight A-fragments (W^T, 16x32 f16 layout) ----
    // A[m][k]: lanes 0-15 hold row m=lane with halves 0..7 = K 0..7
    // (halves 8..15 = K 16..23, lanes 16..31 = K 8..15 -> all zero padding).
    v16h Awi = {};
    v16h Aw[5] = {{}, {}, {}, {}, {}};
    v16h Awo = {};
    if (lane < 8) {
#pragma unroll
        for (int k = 0; k < 3; ++k) Awi[k] = (_Float16)fe_wi[k * 8 + lane];
#pragma unroll
        for (int l = 0; l < 5; ++l)
#pragma unroll
            for (int k = 0; k < 8; ++k)
                Aw[l][k] = (_Float16)fe_w[(l * 8 + k) * 8 + lane];
    }
    if (lane == 0) {
#pragma unroll
        for (int k = 0; k < 8; ++k) Awo[k] = (_Float16)fe_wo[k];
    }

    // ---- hidden-layer biases as WMMA C fragments ----
    // C[m][n]: VGPR r <-> m=r for lanes 0-15 (real rows), m=r+8 (zero pad)
    // for lanes 16-31 -> bias add happens inside the WMMA for free.
    v8f Cb[5];
#pragma unroll
    for (int l = 0; l < 5; ++l)
#pragma unroll
        for (int r = 0; r < 8; ++r)
            Cb[l][r] = (lane < 16) ? fe_b[l * 8 + r] : 0.0f;

    const v8f zeroC = {};

    // ---- bond loop: 32 bonds per wave iteration ----
    float ebond_acc = 0.0f;
    for (int base = wid * 32; base < NBONDS_; base += NWAVES_ * 32) {
        const int j  = base + lane;
        const int i0 = bdid[2 * j + 0];
        const int i1 = bdid[2 * j + 1];

        const float* p0 = x + (size_t)(b * NATOMS_ + i0) * 3;
        const float* p1 = x + (size_t)(b * NATOMS_ + i1) * 3;
        float dx = p0[0] - p1[0];
        float dy = p0[1] - p1[1];
        float dz = p0[2] - p1[2];

        // fractional coords, minimum image, back to cartesian
        float fx = dx * rc[0] + dy * rc[3] + dz * rc[6];
        float fy = dx * rc[1] + dy * rc[4] + dz * rc[7];
        float fz = dx * rc[2] + dy * rc[5] + dz * rc[8];
        fx = (fx > 0.5f) ? fx - 1.0f : fx;  fx = (fx < -0.5f) ? fx + 1.0f : fx;
        fy = (fy > 0.5f) ? fy - 1.0f : fy;  fy = (fy < -0.5f) ? fy + 1.0f : fy;
        fz = (fz > 0.5f) ? fz - 1.0f : fz;  fz = (fz < -0.5f) ? fz + 1.0f : fz;
        float vx = fx * cl[0] + fy * cl[3] + fz * cl[6];
        float vy = fx * cl[1] + fy * cl[4] + fz * cl[7];
        float vz = fx * cl[2] + fy * cl[5] + fz * cl[8];
        float r = sqrtf(vx * vx + vy * vy + vz * vz);

        float e1 = (1.0f + botol) * __expf(bo1 * __powf(r * inv_rosi, bo2));
        float e2 = __expf(bo3 * __powf(r * inv_ropi, bo4));
        float e3 = __expf(bo5 * __powf(r * inv_ropp, bo6));
        float bsi = taperf_(e1, botol, 2.0f * botol) * (e1 - botol);
        float bpi = taperf_(e2, botol, 2.0f * botol) * e2;
        float bpp = taperf_(e3, botol, 2.0f * botol) * e3;

        float bsum = bsi + bpi + bpp;
        float bdpi = bpi + bpp;
        atomicAdd(&sDelta[i0], bsum); atomicAdd(&sDelta[i1], bsum);
        atomicAdd(&sDpi[i0],  bdpi);  atomicAdd(&sDpi[i1],  bdpi);
        atomicAdd(&sSO[i0],   bsi);   atomicAdd(&sSO[i1],   bsi);

        // ---- per-bond MLP: two 16-bond tiles through 7 WMMAs each ----
#pragma unroll
        for (int t = 0; t < 2; ++t) {
            const int src = (lane & 15) + 16 * t;
            float fsi = __shfl(bsi, src);
            float fpi = __shfl(bpi, src);
            float fpp = __shfl(bpp, src);

            // B fragment: activations^T (K=features in halves, N=bond in lane)
            v16h Bin = {};
            if (lane < 16) {
                Bin[0] = (_Float16)fsi;
                Bin[1] = (_Float16)fpi;
                Bin[2] = (_Float16)fpp;
            }
            // input layer (no bias)
            v8f d = __builtin_amdgcn_wmma_f32_16x16x32_f16(
                false, Awi, false, Bin, (short)0, zeroC, false, false);
            v16h act = {};
            if (lane < 16) {
#pragma unroll
                for (int rr = 0; rr < 8; ++rr)
                    act[rr] = (_Float16)sigmoidf_(d[rr]);
            }
            // 5 hidden layers; bias rides the C operand
#pragma unroll
            for (int l = 0; l < 5; ++l) {
                d = __builtin_amdgcn_wmma_f32_16x16x32_f16(
                    false, Aw[l], false, act, (short)0, Cb[l], false, false);
                v16h na = {};
                if (lane < 16) {
#pragma unroll
                    for (int rr = 0; rr < 8; ++rr)
                        na[rr] = (_Float16)sigmoidf_(d[rr]);
                }
                act = na;
            }
            // output layer (8 -> 1)
            d = __builtin_amdgcn_wmma_f32_16x16x32_f16(
                false, Awo, false, act, (short)0, zeroC, false, false);
            if (lane < 16) {
                float esi = sigmoidf_(d[0] + bo_out);
                ebond_acc += -Desi * esi;
            }
        }
    }

    // reduce ebond within wave, accumulate into block energy
#pragma unroll
    for (int off = 16; off > 0; off >>= 1)
        ebond_acc += __shfl_down(ebond_acc, off);
    if (lane == 0) atomicAdd(&sEnergy, ebond_acc);
    __syncthreads();

    // ---- atom-pass uniforms loaded only now (not live across bond loop) ----
    const float lp1 = gp[0], ovun3 = gp[1], ovun4 = gp[2], ovun6 = gp[3];
    const float ovun7 = gp[4], ovun8 = gp[5];

    // ---- per-atom pass ----
    float eatom = 0.0f;
    for (int a = tid; a < NATOMS_; a += THREADS_) {
        float Delta = sDelta[a];
        float Dpi   = sDpi[a];
        float SO    = sSO[a];
        int   s     = spec[a];
        const float* pa = sp_p + s * 5;
        float val = pa[0], vale = pa[1], lp2 = pa[2], ovun2 = pa[3], ovun5 = pa[4];

        float Nlp = 0.5f * (vale - val);
        float de  = 0.5f * (Delta - vale);
        float De  = fminf(ceilf(de), 0.0f);          // -relu(-ceil(de))
        float w   = 1.0f + de - De;
        float nlp = -De + __expf(-lp1 * 4.0f * w * w);
        float Dlp = fmaxf(Nlp - nlp + 1.0f, 0.0f) - 1.0f;
        float Elone = lp2 * Dlp * fast_rcp_(1.0f + __expf(-75.0f * Dlp));

        float dlp = Delta - val - Dlp * fast_rcp_(1.0f + ovun3 * __expf(ovun4 * Dpi));
        float denom = dlp + val;
        float otrm  = fast_rcp_((denom != 0.0f) ? denom : 1e-8f);
        float Eover = SO * otrm * dlp * sigmoidf_(-ovun2 * dlp);
        float Eunder = -ovun5 * (1.0f - __expf(ovun6 * dlp)) * sigmoidf_(ovun2 * dlp)
                       * fast_rcp_(1.0f + ovun7 * __expf(ovun8 * Dpi));
        eatom += Elone + Eover + Eunder;
    }
#pragma unroll
    for (int off = 16; off > 0; off >>= 1)
        eatom += __shfl_down(eatom, off);
    if (lane == 0) atomicAdd(&sEnergy, eatom);
    __syncthreads();

    if (tid == 0) out[b] = sEnergy;
}

extern "C" void kernel_launch(void* const* d_in, const int* in_sizes, int n_in,
                              void* d_out, int out_size, void* d_ws, size_t ws_size,
                              hipStream_t stream) {
    (void)in_sizes; (void)n_in; (void)d_ws; (void)ws_size;
    const float* x     = (const float*)d_in[0];
    const float* cell  = (const float*)d_in[1];
    const float* rcell = (const float*)d_in[2];
    const float* sp_p  = (const float*)d_in[3];
    const float* gp    = (const float*)d_in[4];
    const float* bp    = (const float*)d_in[5];
    const float* fe_wi = (const float*)d_in[6];
    const float* fe_w  = (const float*)d_in[7];
    const float* fe_b  = (const float*)d_in[8];
    const float* fe_wo = (const float*)d_in[9];
    const float* fe_bo = (const float*)d_in[10];
    const int*   bdid  = (const int*)d_in[11];
    const int*   spec  = (const int*)d_in[12];
    float* out = (float*)d_out;

    reaxff_nn_force_kernel<<<dim3(out_size), dim3(THREADS_), 0, stream>>>(
        x, cell, rcell, sp_p, gp, bp, fe_wi, fe_w, fe_b, fe_wo, fe_bo,
        bdid, spec, out);
}
